// GeometryLayer_81028853006804
// MI455X (gfx1250) — matrix operand order repaired and checked by reading, not compiled, not run on hardware
//
#include <hip/hip_runtime.h>

// ---- vector types for WMMA (CDNA5 / gfx1250, wave32) ----
typedef __attribute__((ext_vector_type(16))) __bf16 v16bf;
typedef __attribute__((ext_vector_type(8)))  __bf16 v8bf;
typedef __attribute__((ext_vector_type(4)))  __bf16 v4bf;
typedef __attribute__((ext_vector_type(8)))  float  v8f;
typedef __attribute__((ext_vector_type(4)))  float  v4f;   // native vector (NT-load capable)

// Workload shape (deterministic harness: setup_inputs())
#define NB   8          // batches
#define NL   4800       // tokens per batch (60*80)
#define NC   256        // out channels / feat channels (K after folding)
#define NCIN 320        // conv_w inner dim
#define NM   512        // matches
#define NA   32         // anchors
#define GH   60
#define GW   80

#define BM 128
#define BN 128
#define BK 64
#define LDA 72          // padded LDS stride (bf16 elems): 36 dwords, conflict-free
#define LDB 72

// ---- CDNA5 async global->LDS copy (ASYNCcnt path, §15.18.3 op 98) ----
__device__ __forceinline__ void async_ld_b128(unsigned lds_addr, unsigned goff,
                                              unsigned long long gbase) {
    asm volatile("global_load_async_to_lds_b128 %0, %1, %2"
                 :
                 : "v"(lds_addr), "v"(goff), "s"(gbase)
                 : "memory");
}
__device__ __forceinline__ void wait_async0() {
    asm volatile("s_wait_asynccnt 0x0" ::: "memory");
}

// ---------------------------------------------------------------------------
// 1) Per-batch top-32 of mconf -> anchor coordinates (row,col) for both sides
// ---------------------------------------------------------------------------
__global__ __launch_bounds__(256)
void topk_anchors(const float* __restrict__ mconf,
                  const int* __restrict__ i_ids,
                  const int* __restrict__ j_ids,
                  float* __restrict__ anch)   // [2][NB][NA][2]
{
    __shared__ float sv[NM];
    __shared__ float rv[NM];
    __shared__ int   ri[NM];
    const int b = blockIdx.x, tid = threadIdx.x;

    for (int m = tid; m < NM; m += 256) sv[m] = mconf[b * NM + m];
    __syncthreads();

    for (int a = 0; a < NA; ++a) {
        for (int m = tid; m < NM; m += 256) { rv[m] = sv[m]; ri[m] = m; }
        __syncthreads();
        for (int s = NM >> 1; s > 0; s >>= 1) {
            for (int m = tid; m < s; m += 256) {
                float v2 = rv[m + s]; int i2 = ri[m + s];
                if (v2 > rv[m] || (v2 == rv[m] && i2 < ri[m])) { rv[m] = v2; ri[m] = i2; }
            }
            __syncthreads();
        }
        if (tid == 0) {
            int m = ri[0];
            sv[m] = -3.0e38f;
            int ii = i_ids[b * NM + m];
            int jj = j_ids[b * NM + m];
            int base0 = ((0 * NB + b) * NA + a) * 2;
            int base1 = ((1 * NB + b) * NA + a) * 2;
            anch[base0 + 0] = (float)(ii / GW);
            anch[base0 + 1] = (float)(ii % GW);
            anch[base1 + 0] = (float)(jj / GW);
            anch[base1 + 1] = (float)(jj % GW);
        }
        __syncthreads();
    }
}

// ---------------------------------------------------------------------------
// 2) Convert W0 = conv_w[:, :256] to bf16, packed [256][256]
// ---------------------------------------------------------------------------
__global__ __launch_bounds__(256)
void wconv(const float* __restrict__ w, __bf16* __restrict__ wbf)
{
    const int o = blockIdx.x, k = threadIdx.x;
    wbf[o * NC + k] = (__bf16)w[o * NCIN + k];
}

// ---------------------------------------------------------------------------
// 3) Fold anchors into per-(tensor,batch,channel) bias; column sums of W1.
//    clip is provably inactive: |pos-anchor|/hw < 1 == MAX_COORD_DIST.
// ---------------------------------------------------------------------------
__global__ __launch_bounds__(256)
void geo_bias(const float* __restrict__ w, const float* __restrict__ bias,
              const float* __restrict__ anch,
              float* __restrict__ bias2,   // [2][NB][NC]
              float* __restrict__ wr,      // [2][NC]
              float* __restrict__ wc)      // [2][NC]
{
    const int b = blockIdx.x, t = blockIdx.y, o = threadIdx.x;
    const float invh = 1.0f / (float)GH, invw = 1.0f / (float)GW;
    float acc = bias[o], sR = 0.f, sC = 0.f;
#pragma unroll 4
    for (int a = 0; a < NA; ++a) {
        float wra = w[o * NCIN + NC + 2 * a + 0];
        float wca = w[o * NCIN + NC + 2 * a + 1];
        float ar  = anch[((t * NB + b) * NA + a) * 2 + 0];
        float ac  = anch[((t * NB + b) * NA + a) * 2 + 1];
        acc -= (ar * invh) * wra + (ac * invw) * wca;
        sR += wra; sC += wca;
    }
    bias2[(t * NB + b) * NC + o] = acc;
    if (b == 0) { wr[t * NC + o] = sR * invh; wc[t * NC + o] = sC * invw; }
}

// ---------------------------------------------------------------------------
// 4) Main GEMM: out[t] = bf16(feat[t]) @ bf16(W0)^T  (+ affine geometry, fp32)
//    Block = 128x128 tile, 8 waves (wave32), wave = 2x4 WMMA 16x16 tiles.
//    A: register-staged software pipeline (fp32 stream, fused cvt->LDS).
//    B: global_load_async_to_lds_b128 (ASYNCcnt), L2-resident weight tile.
// ---------------------------------------------------------------------------
__global__ __launch_bounds__(256, 2)
void geo_gemm(const float* __restrict__ feat0, const float* __restrict__ feat1,
              const __bf16* __restrict__ wbf,
              const float* __restrict__ wr_g, const float* __restrict__ wc_g,
              const float* __restrict__ bias2_g, float* __restrict__ out_g)
{
    __shared__ __bf16 Ash[BM * LDA];
    __shared__ __bf16 Bsh[BN * LDB];
    __shared__ float  b2sh[NB * BN];     // bias2 for all batches x block cols
    __shared__ float  wrsh[BN];
    __shared__ float  wcsh[BN];
    __shared__ float  rowRL[BM];
    __shared__ float  rowCL[BM];
    __shared__ int    rowB[BM];          // batch*BN, pre-scaled for b2sh index

    const int t = blockIdx.z;
    const float* __restrict__ feat  = t ? feat1 : feat0;
    const float* __restrict__ wr    = wr_g    + t * NC;
    const float* __restrict__ wc    = wc_g    + t * NC;
    const float* __restrict__ bias2 = bias2_g + t * NB * NC;
    float* __restrict__ out = out_g + (size_t)t * (size_t)NB * NL * NC;

    const int tid  = threadIdx.x;
    const int lane = tid & 31;
    const int wid  = tid >> 5;
    const int wm   = wid >> 1;          // 0..3 : 32-row strip
    const int wn   = wid & 1;           // 0..1 : 64-col strip
    const int nidx = lane & 15;
    const int hi   = lane >> 4;

    const int rowBase = blockIdx.x * BM;
    const int colBase = blockIdx.y * BN;

    // ---- per-thread staging geometry ----
    const int aSeg = tid & 15, aRow = tid >> 4;          // A: seg const, rows +16
    const int bSeg = tid & 7,  bRow = tid >> 3;          // B: seg const, rows +32
    const unsigned ldsBbase = (unsigned)(uintptr_t)(void*)Bsh;
    const unsigned bLdsAddr = ldsBbase + ((unsigned)(bRow * LDB + bSeg * 8)) * 2u;
    const unsigned bGoff    = (unsigned)(bRow * NC + bSeg * 8) * 2u;   // bytes

    // ---- issue async B chunk 0 as early as possible ----
    {
        unsigned long long gbase = (unsigned long long)(uintptr_t)(wbf + (size_t)colBase * NC);
#pragma unroll
        for (int j = 0; j < 4; ++j)
            async_ld_b128(bLdsAddr + (unsigned)(j * 32 * LDB * 2),
                          bGoff + (unsigned)(j * 32 * NC * 2), gbase);
    }

    // ---- prefetch A chunk 0 into registers (non-temporal stream) ----
    v4f areg[8];
#pragma unroll
    for (int j = 0; j < 8; ++j)
        areg[j] = __builtin_nontemporal_load(
            (const v4f*)(feat + (size_t)(rowBase + aRow + 16 * j) * NC + aSeg * 4));

    // ---- preload epilogue tables into LDS ----
    if (tid < BN) {
        wrsh[tid] = wr[colBase + tid];
        wcsh[tid] = wc[colBase + tid];
        int g  = rowBase + tid;
        int b  = g / NL;
        int l  = g - b * NL;
        int rl = l / GW;
        rowB[tid]  = b * BN;
        rowRL[tid] = (float)rl;
        rowCL[tid] = (float)(l - rl * GW);
    }
#pragma unroll
    for (int j = 0; j < 4; ++j) {
        int i = tid + j * 256;                   // NB*BN = 1024 entries
        b2sh[i] = bias2[(i >> 7) * NC + colBase + (i & 127)];
    }

    v8f acc[2][4] = {};

    for (int k0 = 0; k0 < NC; k0 += BK) {
        if (k0 > 0) {
            __syncthreads();       // all waves done reading Ash/Bsh of prev chunk
            // issue async B for this chunk immediately (overlaps A store/prefetch)
            unsigned long long gbase =
                (unsigned long long)(uintptr_t)(wbf + (size_t)colBase * NC + k0);
#pragma unroll
            for (int j = 0; j < 4; ++j)
                async_ld_b128(bLdsAddr + (unsigned)(j * 32 * LDB * 2),
                              bGoff + (unsigned)(j * 32 * NC * 2), gbase);
        }
        // ---- store prefetched A registers as bf16 into LDS ----
#pragma unroll
        for (int j = 0; j < 8; ++j) {
            v4bf h;
            h[0] = (__bf16)areg[j][0]; h[1] = (__bf16)areg[j][1];
            h[2] = (__bf16)areg[j][2]; h[3] = (__bf16)areg[j][3];
            *(v4bf*)(&Ash[(aRow + 16 * j) * LDA + aSeg * 4]) = h;
        }
        // ---- prefetch next A chunk (latency hides under WMMAs below) ----
        if (k0 + BK < NC) {
#pragma unroll
            for (int j = 0; j < 8; ++j)
                areg[j] = __builtin_nontemporal_load(
                    (const v4f*)(feat + (size_t)(rowBase + aRow + 16 * j) * NC
                                 + (k0 + BK) + aSeg * 4));
        }
        wait_async0();             // my B writes to LDS landed
        __syncthreads();           // everyone's A stores + B async visible

        // ---- 2 x K=32 WMMA steps over this chunk ----
#pragma unroll
        for (int ks = 0; ks < BK; ks += 32) {
            v16bf afr[2], bfr[4];
            // A fragment (16-bit A 16x32: e<8 -> K=hi*8+e, e>=8 -> K=16+hi*8+e-8)
#pragma unroll
            for (int tm = 0; tm < 2; ++tm) {
                int r = wm * 32 + tm * 16 + nidx;
                const v8bf lo = *(const v8bf*)(&Ash[r * LDA + ks + hi * 8]);
                const v8bf hh = *(const v8bf*)(&Ash[r * LDA + ks + hi * 8 + 16]);
                afr[tm] = __builtin_shufflevector(lo, hh,
                            0,1,2,3,4,5,6,7,8,9,10,11,12,13,14,15);
            }
            // B fragment (col = lane&15, e<8 -> K=hi*8+e, e>=8 -> K=16+hi*8+e-8)
#pragma unroll
            for (int tn = 0; tn < 4; ++tn) {
                int c = wn * 64 + tn * 16 + nidx;
                const v8bf lo = *(const v8bf*)(&Bsh[c * LDB + ks + hi * 8]);
                const v8bf hh = *(const v8bf*)(&Bsh[c * LDB + ks + hi * 8 + 16]);
                bfr[tn] = __builtin_shufflevector(lo, hh,
                            0,1,2,3,4,5,6,7,8,9,10,11,12,13,14,15);
            }
#pragma unroll
            for (int tm = 0; tm < 2; ++tm)
#pragma unroll
                for (int tn = 0; tn < 4; ++tn)
                    acc[tm][tn] = __builtin_amdgcn_wmma_f32_16x16x32_bf16(
                        false, afr[tm], false, bfr[tn],
                        (short)0, acc[tm][tn], false, false);
        }
    }

    // ---- epilogue: + row*wr + col*wc + bias2[b], all LDS-served, NT stores ----
#pragma unroll
    for (int tm = 0; tm < 2; ++tm) {
        const int rloc  = wm * 32 + tm * 16 + hi * 8;
        const int rbase = rowBase + rloc;
#pragma unroll
        for (int tn = 0; tn < 4; ++tn) {
            const int cloc = wn * 64 + tn * 16 + nidx;
            const int c    = colBase + cloc;
            const float wrv = wrsh[cloc];
            const float wcv = wcsh[cloc];
#pragma unroll
            for (int v = 0; v < 8; ++v) {
                const int rl = rloc + v;
                const float g = rowRL[rl] * wrv + rowCL[rl] * wcv
                              + b2sh[rowB[rl] + cloc];
                __builtin_nontemporal_store(acc[tm][tn][v] + g,
                                            &out[(size_t)(rbase + v) * NC + c]);
            }
        }
    }
}

// ---------------------------------------------------------------------------
// Launch
// ---------------------------------------------------------------------------
extern "C" void kernel_launch(void* const* d_in, const int* in_sizes, int n_in,
                              void* d_out, int out_size, void* d_ws, size_t ws_size,
                              hipStream_t stream) {
    const float* feat0  = (const float*)d_in[0];
    const float* feat1  = (const float*)d_in[1];
    const float* mconf  = (const float*)d_in[2];
    const float* conv_w = (const float*)d_in[3];
    const float* conv_b = (const float*)d_in[4];
    const int*   i_ids  = (const int*)d_in[5];
    const int*   j_ids  = (const int*)d_in[6];
    float* out = (float*)d_out;

    char* ws = (char*)d_ws;
    float*  anch  = (float*)(ws);                         //  2*8*32*2 = 1024 f32
    float*  bias2 = (float*)(ws + 4096);                  //  2*8*256  = 4096 f32
    float*  wr    = (float*)(ws + 4096 + 16384);          //  2*256 f32
    float*  wc    = (float*)(ws + 4096 + 16384 + 2048);   //  2*256 f32
    __bf16* wbf   = (__bf16*)(ws + 4096 + 16384 + 4096);  //  256*256 bf16

    topk_anchors<<<NB, 256, 0, stream>>>(mconf, i_ids, j_ids, anch);
    wconv<<<NC, 256, 0, stream>>>(conv_w, wbf);
    geo_bias<<<dim3(NB, 2), 256, 0, stream>>>(conv_w, conv_b, anch, bias2, wr, wc);

    dim3 grid((NB * NL) / BM, NC / BN, 2);   // (300, 2, 2)
    geo_gemm<<<grid, 256, 0, stream>>>(feat0, feat1, wbf, wr, wc, bias2, out);
}
